// GPT_44117904064902
// MI455X (gfx1250) — matrix operand order repaired and checked by reading, not compiled, not run on hardware
//
#include <hip/hip_runtime.h>
#include <hip/hip_bf16.h>
#include <math.h>

// ---------------------------------------------------------------------------
// Types for CDNA5 WMMA
// ---------------------------------------------------------------------------
typedef __bf16 bf16_t;
typedef __attribute__((ext_vector_type(16))) __bf16 v16bf;
typedef __attribute__((ext_vector_type(8)))  __bf16 v8bf;
typedef __attribute__((ext_vector_type(8)))  float  v8f;

#define D_MODEL   1024
#define N_LAYERS  12
#define BATCH     256
#define SEQ       64
#define TOKENS    (BATCH * SEQ)   // 16384
#define N_HEADS   16
#define HEAD_DIM  64

__device__ __forceinline__ bf16_t f2bf(float f) {
    unsigned u = __builtin_bit_cast(unsigned, f);
    unsigned r = u + 0x7fffu + ((u >> 16) & 1u);   // round-to-nearest-even
    unsigned short h = (unsigned short)(r >> 16);
    return __builtin_bit_cast(bf16_t, h);
}
__device__ __forceinline__ float bf2f(bf16_t b) {
    unsigned short h = __builtin_bit_cast(unsigned short, b);
    unsigned u = ((unsigned)h) << 16;
    return __builtin_bit_cast(float, u);
}

// A fragment: lane half h holds K = 16h + j (contiguous 32B per lane)
__device__ __forceinline__ v16bf load_a_frag(const bf16_t* p) {
    v8bf lo = *(const v8bf*)(p);
    v8bf hi = *(const v8bf*)(p + 8);
    return __builtin_shufflevector(lo, hi,
        0, 1, 2, 3, 4, 5, 6, 7, 8, 9, 10, 11, 12, 13, 14, 15);
}

// ---------------------------------------------------------------------------
// Pack a [K,N] f32 weight into bf16 WMMA B-fragments.
// Fragment layout: frag (n_tile, k_tile); within a fragment, lane L holds
// 16 contiguous bf16: N = n_tile*16 + (L&15), K = k_tile*32 + (L>>4)*16 + j.
// Per-lane 32B contiguous -> wave B-frag load is 1KB fully coalesced.
// ---------------------------------------------------------------------------
__global__ __launch_bounds__(256) void pack_b_kernel(
    const float* __restrict__ W, bf16_t* __restrict__ out, int K, int N)
{
    long long idx = (long long)blockIdx.x * 256 + threadIdx.x;
    long long total = (long long)K * N;
    if (idx >= total) return;
    int k = (int)(idx / N);
    int n = (int)(idx % N);
    int kt = k >> 5, kk = k & 31, nt = n >> 4;
    int lane = (n & 15) | (((kk >> 4) & 1) << 4);
    int j = kk & 15;
    int Kt = K >> 5;
    size_t off = (((size_t)nt * Kt + kt) * 32 + lane) * 16 + j;
    out[off] = f2bf(W[idx]);
}

// ---------------------------------------------------------------------------
// Embedding: x[row] = tok_emb[idx[row]] + pos_emb[row % SEQ]   (f32 + bf16)
// ---------------------------------------------------------------------------
__global__ __launch_bounds__(256) void embed_kernel(
    const int* __restrict__ idx, const float* __restrict__ tok,
    const float* __restrict__ pos, float* __restrict__ x,
    bf16_t* __restrict__ xb)
{
    int row = blockIdx.x;
    int t = idx[row];
    int s = row & (SEQ - 1);
    const float* pt = tok + (size_t)t * D_MODEL;
    const float* pp = pos + (size_t)s * D_MODEL;
    for (int c = threadIdx.x; c < D_MODEL; c += 256) {
        float v = pt[c] + pp[c];
        x [(size_t)row * D_MODEL + c] = v;
        xb[(size_t)row * D_MODEL + c] = f2bf(v);
    }
}

// ---------------------------------------------------------------------------
// bf16 WMMA GEMM: C[M,N] = A[M,K] @ Bpacked + bias (+resid) (opt GELU)
// Block = 256 threads = 8 waves arranged 4(M) x 2(N): block tile 128x128.
// Each wave owns a 32x64 tile = 2x4 WMMA tiles, register double-buffered
// over k-tiles so loads for kt+1 overlap the 8 WMMAs of kt.
// ---------------------------------------------------------------------------
__global__ __launch_bounds__(256) void gemm_bf16_kernel(
    const bf16_t* __restrict__ A, const bf16_t* __restrict__ Bp,
    const float* __restrict__ bias, const float* __restrict__ resid,
    float* __restrict__ outF, bf16_t* __restrict__ outB,
    int M, int N, int K, int fuse_gelu)
{
    const int lane = threadIdx.x & 31;
    const int wave = threadIdx.x >> 5;
    const int wy = wave >> 1;          // 0..3 (M direction)
    const int wx = wave & 1;           // 0..1 (N direction)
    const int m_wave = blockIdx.y * 128 + wy * 32;
    const int n_wave = blockIdx.x * 128 + wx * 64;
    const int Kt = K >> 5;
    const int half = lane >> 4;

    // A pointers: row0 = m_wave + (lane&15), row1 = row0 + 16
    const bf16_t* pa0 = A + (size_t)(m_wave + (lane & 15)) * K + half * 16;
    // Packed B pointer for n_tile base of this wave
    const size_t nstride = (size_t)Kt * 512;  // one n_tile step in packed B
    const bf16_t* pb = Bp + (((size_t)(n_wave >> 4) * Kt) * 32 + lane) * 16;

    v8f acc[2][4];
#pragma unroll
    for (int i = 0; i < 2; ++i)
#pragma unroll
        for (int t = 0; t < 4; ++t) acc[i][t] = (v8f){};

    v16bf aC[2], bC[4], aN[2], bN[4];

    auto load_tile = [&](int kt, v16bf (&a)[2], v16bf (&b)[4]) {
        const bf16_t* p = pa0 + (size_t)kt * 32;
        a[0] = load_a_frag(p);
        a[1] = load_a_frag(p + (size_t)16 * K);
        const bf16_t* q = pb + (size_t)kt * 512;
        b[0] = *(const v16bf*)(q);
        b[1] = *(const v16bf*)(q + nstride);
        b[2] = *(const v16bf*)(q + 2 * nstride);
        b[3] = *(const v16bf*)(q + 3 * nstride);
    };
    auto compute = [&](v16bf (&a)[2], v16bf (&b)[4]) {
#pragma unroll
        for (int i = 0; i < 2; ++i)
#pragma unroll
            for (int t = 0; t < 4; ++t)
                acc[i][t] = __builtin_amdgcn_wmma_f32_16x16x32_bf16(
                    false, a[i], false, b[t], (short)0, acc[i][t], false, false);
    };

    // Kt is always even (K = 1024 or 4096): ping-pong double buffer.
    load_tile(0, aC, bC);
    for (int kt = 0; kt < Kt; kt += 2) {
        load_tile(kt + 1, aN, bN);      // prefetch while computing C
        compute(aC, bC);
        if (kt + 2 < Kt) load_tile(kt + 2, aC, bC);
        compute(aN, bN);
    }

    // Epilogue. C/D layout: elem r of lane: M = m + half*8 + r,
    //                                       N = n_wave + t*16 + (lane&15)
    const int ncol0 = n_wave + (lane & 15);
#pragma unroll
    for (int i = 0; i < 2; ++i) {
        const int mrow0 = m_wave + i * 16 + half * 8;
#pragma unroll
        for (int t = 0; t < 4; ++t) {
            int n = ncol0 + t * 16;
            float bn = bias[n];
#pragma unroll
            for (int r = 0; r < 8; ++r) {
                int m = mrow0 + r;
                size_t o = (size_t)m * N + n;
                float v = acc[i][t][r] + bn;
                if (resid) v += resid[o];
                if (fuse_gelu) v = 0.5f * v * (1.0f + erff(v * 0.70710678118654752f));
                if (outF) outF[o] = v;
                if (outB) outB[o] = f2bf(v);
            }
        }
    }
}

// ---------------------------------------------------------------------------
// Attention: one 64-thread block per (batch, head). Reproduces reference
// quirks: scale 1/sqrt(d_model)=1/32, softmax over the QUERY axis per key.
// qs is reused for the attention matrix after phase 1.
// ---------------------------------------------------------------------------
__global__ __launch_bounds__(64) void attn_kernel(
    const bf16_t* __restrict__ qkv, bf16_t* __restrict__ y)
{
    __shared__ float qs[SEQ][HEAD_DIM + 1];  // later reused as attn[i][j]
    __shared__ float ks[SEQ][HEAD_DIM + 1];
    __shared__ float vs[SEQ][HEAD_DIM + 1];

    const int b = blockIdx.x >> 4;
    const int h = blockIdx.x & 15;
    const int tid = threadIdx.x;  // 0..63

    const bf16_t* prow = qkv + ((size_t)(b * SEQ + tid)) * (3 * D_MODEL)
                             + h * HEAD_DIM;
#pragma unroll 4
    for (int c = 0; c < HEAD_DIM; ++c) {
        qs[tid][c] = bf2f(prow[c]);
        ks[tid][c] = bf2f(prow[D_MODEL + c]);
        vs[tid][c] = bf2f(prow[2 * D_MODEL + c]);
    }
    __syncthreads();

    // phase 1: thread j computes logits column j, softmax over i>=j
    const int j = tid;
    float logit[SEQ];
    for (int i = 0; i < SEQ; ++i) {
        float s = 0.0f;
#pragma unroll 8
        for (int c = 0; c < HEAD_DIM; ++c) s += qs[i][c] * ks[j][c];
        logit[i] = s * (1.0f / 32.0f);  // 1/sqrt(d_model)
    }
    float mx = -3.0e38f;
    for (int i = j; i < SEQ; ++i) mx = fmaxf(mx, logit[i]);
    float den = 0.0f;
    for (int i = j; i < SEQ; ++i) { float e = __expf(logit[i] - mx); logit[i] = e; den += e; }
    float inv = 1.0f / den;
    __syncthreads();                    // everyone done reading qs
    for (int i = 0; i < SEQ; ++i) qs[i][j] = (i >= j) ? logit[i] * inv : 0.0f;
    __syncthreads();

    // phase 2: thread c computes output column c: y[i][c] = sum_j a[i][j]*v[j][c]
    const int c = tid;
    for (int i = 0; i < SEQ; ++i) {
        float s = 0.0f;
#pragma unroll 8
        for (int jj = 0; jj < SEQ; ++jj) s += qs[i][jj] * vs[jj][c];
        y[((size_t)(b * SEQ + i)) * D_MODEL + h * HEAD_DIM + c] = f2bf(s);
    }
}

// ---------------------------------------------------------------------------
// LayerNorm over last dim (1024), one block per row, bf16 output for WMMA A.
// ---------------------------------------------------------------------------
__global__ __launch_bounds__(256) void ln_kernel(
    const float* __restrict__ x, const float* __restrict__ g,
    const float* __restrict__ bb, bf16_t* __restrict__ out)
{
    __shared__ float red[256];
    const int row = blockIdx.x;
    const float* px = x + (size_t)row * D_MODEL;
    float lsum = 0.0f, lsq = 0.0f;
    for (int c = threadIdx.x; c < D_MODEL; c += 256) {
        float v = px[c];
        lsum += v; lsq += v * v;
    }
    red[threadIdx.x] = lsum;
    __syncthreads();
    for (int s = 128; s > 0; s >>= 1) {
        if (threadIdx.x < s) red[threadIdx.x] += red[threadIdx.x + s];
        __syncthreads();
    }
    float mean = red[0] * (1.0f / D_MODEL);
    __syncthreads();
    red[threadIdx.x] = lsq;
    __syncthreads();
    for (int s = 128; s > 0; s >>= 1) {
        if (threadIdx.x < s) red[threadIdx.x] += red[threadIdx.x + s];
        __syncthreads();
    }
    float var = red[0] * (1.0f / D_MODEL) - mean * mean;
    float rstd = rsqrtf(var + 1e-5f);
    for (int c = threadIdx.x; c < D_MODEL; c += 256) {
        float v = (px[c] - mean) * rstd * g[c] + bb[c];
        out[(size_t)row * D_MODEL + c] = f2bf(v);
    }
}

// ---------------------------------------------------------------------------
// Host side
// ---------------------------------------------------------------------------
extern "C" void kernel_launch(void* const* d_in, const int* in_sizes, int n_in,
                              void* d_out, int out_size, void* d_ws, size_t ws_size,
                              hipStream_t stream)
{
    (void)in_sizes; (void)n_in; (void)out_size; (void)ws_size;

    const int*   idx   = (const int*)  d_in[0];
    const float* tok   = (const float*)d_in[1];
    const float* pos   = (const float*)d_in[2];
    const float* wqkv  = (const float*)d_in[3];
    const float* bqkv  = (const float*)d_in[4];
    const float* wo    = (const float*)d_in[5];
    const float* bo    = (const float*)d_in[6];
    const float* w1    = (const float*)d_in[7];
    const float* b1    = (const float*)d_in[8];
    const float* w2    = (const float*)d_in[9];
    const float* b2    = (const float*)d_in[10];
    const float* ln_g  = (const float*)d_in[11];
    const float* ln_b  = (const float*)d_in[12];
    const float* out_w = (const float*)d_in[13];
    const float* out_b = (const float*)d_in[14];

    // -------- workspace carve-out --------
    char* ws = (char*)d_ws;
    size_t off = 0;
    auto carve = [&](size_t bytes) -> void* {
        off = (off + 255) & ~(size_t)255;
        void* p = ws + off;
        off += bytes;
        return p;
    };
    const size_t WQKV_L = (size_t)D_MODEL * 3 * D_MODEL;      // 3,145,728
    const size_t WO_L   = (size_t)D_MODEL * D_MODEL;          // 1,048,576
    const size_t W1_L   = (size_t)D_MODEL * 4 * D_MODEL;      // 4,194,304
    const size_t W2_L   = (size_t)4 * D_MODEL * D_MODEL;      // 4,194,304

    bf16_t* pk_wqkv = (bf16_t*)carve(N_LAYERS * WQKV_L * 2);
    bf16_t* pk_wo   = (bf16_t*)carve(N_LAYERS * WO_L   * 2);
    bf16_t* pk_w1   = (bf16_t*)carve(N_LAYERS * W1_L   * 2);
    bf16_t* pk_w2   = (bf16_t*)carve(N_LAYERS * W2_L   * 2);
    bf16_t* pk_ow   = (bf16_t*)carve(WO_L * 2);
    float*  x       = (float*) carve((size_t)TOKENS * D_MODEL * 4);
    bf16_t* xb      = (bf16_t*)carve((size_t)TOKENS * D_MODEL * 2);
    bf16_t* qkvb    = (bf16_t*)carve((size_t)TOKENS * 3 * D_MODEL * 2);
    bf16_t* yb      = (bf16_t*)carve((size_t)TOKENS * D_MODEL * 2);
    bf16_t* hb      = (bf16_t*)carve((size_t)TOKENS * 4 * D_MODEL * 2);

    auto cdiv = [](size_t a, size_t b) { return (unsigned)((a + b - 1) / b); };

    // -------- pack all weights to bf16 WMMA B-fragment layout --------
    for (int l = 0; l < N_LAYERS; ++l) {
        pack_b_kernel<<<cdiv(WQKV_L, 256), 256, 0, stream>>>(
            wqkv + (size_t)l * WQKV_L, pk_wqkv + (size_t)l * WQKV_L,
            D_MODEL, 3 * D_MODEL);
        pack_b_kernel<<<cdiv(WO_L, 256), 256, 0, stream>>>(
            wo + (size_t)l * WO_L, pk_wo + (size_t)l * WO_L,
            D_MODEL, D_MODEL);
        pack_b_kernel<<<cdiv(W1_L, 256), 256, 0, stream>>>(
            w1 + (size_t)l * W1_L, pk_w1 + (size_t)l * W1_L,
            D_MODEL, 4 * D_MODEL);
        pack_b_kernel<<<cdiv(W2_L, 256), 256, 0, stream>>>(
            w2 + (size_t)l * W2_L, pk_w2 + (size_t)l * W2_L,
            4 * D_MODEL, D_MODEL);
    }
    pack_b_kernel<<<cdiv(WO_L, 256), 256, 0, stream>>>(
        out_w, pk_ow, D_MODEL, D_MODEL);

    // -------- embedding --------
    embed_kernel<<<TOKENS, 256, 0, stream>>>(idx, tok, pos, x, xb);

    // -------- transformer layers --------
    const int M = TOKENS;                       // 16384
    for (int l = 0; l < N_LAYERS; ++l) {
        // qkv = xb @ wqkv + bqkv  -> bf16 qkvb
        gemm_bf16_kernel<<<dim3((3 * D_MODEL) / 128, M / 128), 256, 0, stream>>>(
            xb, pk_wqkv + (size_t)l * WQKV_L, bqkv + (size_t)l * 3 * D_MODEL,
            nullptr, nullptr, qkvb, M, 3 * D_MODEL, D_MODEL, 0);

        // attention -> bf16 yb
        attn_kernel<<<BATCH * N_HEADS, 64, 0, stream>>>(qkvb, yb);

        // x = yb @ wo + bo + x   (f32 in place) ; xb = bf16(x)
        gemm_bf16_kernel<<<dim3(D_MODEL / 128, M / 128), 256, 0, stream>>>(
            yb, pk_wo + (size_t)l * WO_L, bo + (size_t)l * D_MODEL,
            x, x, xb, M, D_MODEL, D_MODEL, 0);

        // h = gelu(xb @ w1 + b1) -> bf16 hb
        gemm_bf16_kernel<<<dim3((4 * D_MODEL) / 128, M / 128), 256, 0, stream>>>(
            xb, pk_w1 + (size_t)l * W1_L, b1 + (size_t)l * 4 * D_MODEL,
            nullptr, nullptr, hb, M, 4 * D_MODEL, D_MODEL, 1);

        // x = hb @ w2 + b2 + x ; xb = bf16(x)
        gemm_bf16_kernel<<<dim3(D_MODEL / 128, M / 128), 256, 0, stream>>>(
            hb, pk_w2 + (size_t)l * W2_L, b2 + (size_t)l * D_MODEL,
            x, x, xb, M, D_MODEL, 4 * D_MODEL, 0);
    }

    // -------- final layernorm + projection --------
    ln_kernel<<<TOKENS, 256, 0, stream>>>(x, ln_g, ln_b, xb);
    gemm_bf16_kernel<<<dim3(D_MODEL / 128, M / 128), 256, 0, stream>>>(
        xb, pk_ow, out_b, nullptr, (float*)d_out, nullptr,
        M, D_MODEL, D_MODEL, 0);
}
